// GRU_67851893342562
// MI455X (gfx1250) — compile-verified, matrix operand order checked by reading, not run on hardware
//
#include <hip/hip_runtime.h>

// ---------------------------------------------------------------------------
// Types for CDNA5 WMMA (wave32): v_wmma_f32_16x16x32_bf16
// ---------------------------------------------------------------------------
typedef __attribute__((ext_vector_type(16))) __bf16 v16bf;
typedef __attribute__((ext_vector_type(8)))  float  v8f;

#define WMMA_BF16(a, b, c) \
  __builtin_amdgcn_wmma_f32_16x16x32_bf16(false, (a), false, (b), (short)0, (c), false, false)

// ---------------------------------------------------------------------------
// CDNA5 async global->LDS copy (16 bytes per lane), tracked by ASYNCcnt.
// Builtin signature (from hipcc diagnostic): (v4i addrspace(1)*, v4i addrspace(3)*, Imm, Imm)
// ---------------------------------------------------------------------------
typedef int v4i_async __attribute__((vector_size(16)));

#if __has_builtin(__builtin_amdgcn_global_load_async_to_lds_b128)
#define ASYNC_CP16(dst, src)                                              \
  __builtin_amdgcn_global_load_async_to_lds_b128(                         \
      (__attribute__((address_space(1))) v4i_async*)(src),                \
      (__attribute__((address_space(3))) v4i_async*)(dst), 0, 0)
#else
#define ASYNC_CP16(dst, src)                                              \
  asm volatile("global_load_async_to_lds_b128 %0, %1, off" ::             \
               "v"((__attribute__((address_space(3))) v4i_async*)(dst)),  \
               "v"((const void*)(src))                                    \
               : "memory")
#endif

__device__ __forceinline__ void wait_async0() {
#if __has_builtin(__builtin_amdgcn_s_wait_asynccnt)
  __builtin_amdgcn_s_wait_asynccnt(0);
#else
  asm volatile("s_wait_asynccnt 0x0" ::: "memory");
#endif
}

__device__ __forceinline__ unsigned short f2bf(float f) {
  union { float f; unsigned int u; } x; x.f = f;
  unsigned int u = x.u;
  unsigned int r = (u + 0x7FFFu + ((u >> 16) & 1u)) >> 16; // round-to-nearest-even
  return (unsigned short)r;
}

// Load a 16x32 bf16 WMMA A/B fragment. Source rows are contiguous (stride `ld`
// elements). Lane l holds row (l&15); K-chunks [(l>>4)*8, +8) and +16 more.
// Works for global pointers and pointers into __shared__ tiles (ds loads).
template <typename PT>
__device__ __forceinline__ v16bf load_frag(PT base, int ld, int lane) {
  const int r  = lane & 15;
  const int kh = (lane >> 4) * 8;
  PT p = base + (size_t)r * ld + kh;
  union { v16bf v; uint4 u[2]; } un;
  un.u[0] = *(const uint4*)(p);
  un.u[1] = *(const uint4*)(p + 16);
  return un.v;
}

// ---------------------------------------------------------------------------
// Elementwise helpers
// ---------------------------------------------------------------------------
__global__ void cvt_f32_bf16_kernel(const float* __restrict__ src,
                                    unsigned short* __restrict__ dst, size_t n) {
  size_t i = (size_t)blockIdx.x * blockDim.x + threadIdx.x;
  size_t stride = (size_t)gridDim.x * blockDim.x;
  for (; i < n; i += stride) dst[i] = f2bf(src[i]);
}

__global__ void zero_h_kernel(float* __restrict__ hf,
                              unsigned short* __restrict__ hb, int n) {
  int i = blockIdx.x * blockDim.x + threadIdx.x;
  if (i < n) { hf[i] = 0.0f; hb[i] = 0; }
}

__global__ void copy_f32_kernel(const float* __restrict__ src,
                                float* __restrict__ dst, int n) {
  int i = blockIdx.x * blockDim.x + threadIdx.x;
  if (i < n) dst[i] = src[i];
}

// ---------------------------------------------------------------------------
// Big GEMM: C[M,N] = A[M,K](bf16) * Bw[N,K](bf16)^T + bias[N]   (fp32 out)
// Block tile 128x128, K-step 32, 256 threads = 8 waves (2 Mwaves x 4 Nwaves),
// each wave: 4x2 = 8 WMMA accumulators (64x32 output).
// Double-buffered LDS tiles filled with ASYNC global->LDS copies:
//   per iteration: s_wait_asynccnt 0 + one barrier, then stage next tile
//   (into the buffer everyone finished reading before this barrier) while
//   the WMMAs consume the current tile.
// ---------------------------------------------------------------------------
#define BM 128
#define BN 128
#define BKK 32

__global__ void __launch_bounds__(256)
gemm_bias_kernel(const unsigned short* __restrict__ A,
                 const unsigned short* __restrict__ Bw,
                 const float* __restrict__ bias,
                 float* __restrict__ C,
                 int M, int N, int K) {
  __shared__ __align__(16) unsigned short As[2][BM * BKK];
  __shared__ __align__(16) unsigned short Bs[2][BN * BKK];

  const int tid  = threadIdx.x;
  const int lane = tid & 31;
  const int wid  = tid >> 5;
  const int bm = blockIdx.x * BM;
  const int bn = blockIdx.y * BN;
  const int wm = (wid >> 2) * 64;   // 0 or 64
  const int wn = (wid & 3) * 32;    // 0,32,64,96

  v8f acc[4][2] = {};

  // Stage one 128x32 A tile + one 128x32 B tile into LDS buffer `buf`.
  // 512 16B-chunks per tile, 256 threads -> 2 chunks each, coalesced.
  auto stage = [&](int buf, int k0) {
#pragma unroll
    for (int c = 0; c < 2; ++c) {
      const int lin = c * 256 + tid;   // chunk id 0..511
      const int row = lin >> 2;        // 4 chunks of 8 elems per 32-elem row
      const int col = (lin & 3) * 8;
      ASYNC_CP16(&As[buf][row * BKK + col], &A[(size_t)(bm + row) * K + k0 + col]);
      ASYNC_CP16(&Bs[buf][row * BKK + col], &Bw[(size_t)(bn + row) * K + k0 + col]);
    }
  };

  stage(0, 0);
  const int nk = K / BKK;
  for (int ki = 0; ki < nk; ++ki) {
    const int cur = ki & 1;
    wait_async0();        // my async writes into buf `cur` have landed
    __syncthreads();      // everyone's writes landed; prev reads of buf cur^1 done
    if (ki + 1 < nk) stage(cur ^ 1, (ki + 1) * BKK);

    v16bf bf[2];
#pragma unroll
    for (int nt = 0; nt < 2; ++nt)
      bf[nt] = load_frag(&Bs[cur][(wn + nt * 16) * BKK], BKK, lane);
#pragma unroll
    for (int mt = 0; mt < 4; ++mt) {
      v16bf af = load_frag(&As[cur][(wm + mt * 16) * BKK], BKK, lane);
#pragma unroll
      for (int nt = 0; nt < 2; ++nt)
        acc[mt][nt] = WMMA_BF16(af, bf[nt], acc[mt][nt]);
    }
    // no trailing barrier: next iteration's wait+barrier provides the fence
  }

  // Epilogue: C element (m,n) = acc[..][..][v],  m = base+v+8*(lane>>4), n = base+(lane&15)
  const int nlane = lane & 15;
  const int moff  = (lane >> 4) * 8;
#pragma unroll
  for (int mt = 0; mt < 4; ++mt) {
#pragma unroll
    for (int nt = 0; nt < 2; ++nt) {
      const int ng = bn + wn + nt * 16 + nlane;
      const float bv = bias[ng];
#pragma unroll
      for (int v = 0; v < 8; ++v) {
        const int mg = bm + wm + mt * 16 + v + moff;
        C[(size_t)mg * N + ng] = acc[mt][nt][v] + bv;
      }
    }
  }
}

// ---------------------------------------------------------------------------
// One GRU timestep. Each workgroup owns a 16-wide j-strip of H and computes
// the r/z/n gate strips (3 GEMM column-strips of [64 x 16], K = H) so it can
// finish the gate nonlinearity locally. 128 threads = 4 waves; wave w handles
// batch rows [16w, 16w+16). h double-buffered (h_in/h_out) across steps.
// Gate inputs (gx rows, h_in) are prefetched up front so their HBM/L2 latency
// hides behind the 32-iteration recurrent GEMM (step kernels are latency-bound).
// ---------------------------------------------------------------------------
__global__ void __launch_bounds__(128)
gru_step_kernel(const unsigned short* __restrict__ hb_in,  // [B,H] bf16
                const unsigned short* __restrict__ Whh,    // [3H,H] bf16
                const float* __restrict__ gx,              // [B*T,3H] fp32 (+b_ih)
                const float* __restrict__ b_hh,            // [3H]
                const float* __restrict__ h_in,            // [B,H] fp32
                float* __restrict__ h_out,                 // [B,H] fp32
                unsigned short* __restrict__ hb_out,       // [B,H] bf16
                unsigned short* __restrict__ y_bf,         // [B,T,H] bf16 or null
                float* __restrict__ y_f,                   // [B,T,H] fp32 or null
                int t, int T, int H) {
  const int lane = threadIdx.x & 31;
  const int w    = threadIdx.x >> 5;
  const int jb   = blockIdx.x * 16;   // j-strip base within H
  const int mrow = w * 16;            // batch-row base
  const int n    = lane & 15;
  const int j    = jb + n;
  const int moff = (lane >> 4) * 8;

  // prefetch gate inputs (overlaps with GEMM below) -> global_prefetch_b8
#pragma unroll
  for (int v = 0; v < 8; ++v) {
    const int b = mrow + v + moff;
    const float* gp = gx + ((size_t)b * T + t) * (size_t)(3 * H) + j;
    __builtin_prefetch(gp, 0, 0);
    __builtin_prefetch(gp + H, 0, 0);
    __builtin_prefetch(gp + 2 * H, 0, 0);
    __builtin_prefetch(&h_in[(size_t)b * H + j], 0, 0);
  }

  v8f acc[3] = {};
  for (int k0 = 0; k0 < H; k0 += 32) {
    v16bf af = load_frag(&hb_in[(size_t)mrow * H + k0], H, lane);
#pragma unroll
    for (int g = 0; g < 3; ++g) {
      v16bf bf = load_frag(&Whh[(size_t)(g * H + jb) * H + k0], H, lane);
      acc[g] = WMMA_BF16(af, bf, acc[g]);
    }
  }

  const float bhr = b_hh[j];
  const float bhz = b_hh[H + j];
  const float bhn = b_hh[2 * H + j];

#pragma unroll
  for (int v = 0; v < 8; ++v) {
    const int b = mrow + v + moff;
    const size_t gxrow = ((size_t)b * T + t) * (size_t)(3 * H);
    const float xr = gx[gxrow + j];
    const float xz = gx[gxrow + H + j];
    const float xn = gx[gxrow + 2 * H + j];
    const float hr = acc[0][v] + bhr;
    const float hz = acc[1][v] + bhz;
    const float hn = acc[2][v] + bhn;
    const float r  = 1.0f / (1.0f + __expf(-(xr + hr)));
    const float z  = 1.0f / (1.0f + __expf(-(xz + hz)));
    const float nn = tanhf(xn + r * hn);
    const float hp = h_in[(size_t)b * H + j];
    const float hnew = (1.0f - z) * nn + z * hp;
    h_out[(size_t)b * H + j] = hnew;
    hb_out[(size_t)b * H + j] = f2bf(hnew);
    const size_t yi = ((size_t)b * T + t) * (size_t)H + j;
    if (y_bf) y_bf[yi] = f2bf(hnew);
    if (y_f)  y_f[yi]  = hnew;
  }
}

// ---------------------------------------------------------------------------
// Host orchestration
// ---------------------------------------------------------------------------
extern "C" void kernel_launch(void* const* d_in, const int* in_sizes, int n_in,
                              void* d_out, int out_size, void* d_ws, size_t ws_size,
                              hipStream_t stream) {
  (void)in_sizes; (void)n_in; (void)out_size; (void)ws_size;
  const int B = 64, T = 512, D = 1024, H = 1024;
  const size_t MT = (size_t)B * T;        // 32768
  const int N3 = 3 * H;                   // 3072

  const float* x     = (const float*)d_in[0];
  const float* wih0  = (const float*)d_in[1];
  const float* whh0  = (const float*)d_in[2];
  const float* bih0  = (const float*)d_in[3];
  const float* bhh0  = (const float*)d_in[4];
  const float* wih1  = (const float*)d_in[5];
  const float* whh1  = (const float*)d_in[6];
  const float* bih1  = (const float*)d_in[7];
  const float* bhh1  = (const float*)d_in[8];
  float* out = (float*)d_out;             // [B*T*H] y1, then [B*H] h0, [B*H] h1

  char* ws = (char*)d_ws;
  size_t off = 0;
  auto take = [&](size_t bytes) -> char* {
    char* p = ws + off;
    off = (off + bytes + 255) & ~(size_t)255;
    return p;
  };
  unsigned short* xb    = (unsigned short*)take(MT * D * 2);
  unsigned short* y0b   = (unsigned short*)take(MT * H * 2);
  unsigned short* wbih0 = (unsigned short*)take((size_t)N3 * D * 2);
  unsigned short* wbhh0 = (unsigned short*)take((size_t)N3 * H * 2);
  unsigned short* wbih1 = (unsigned short*)take((size_t)N3 * H * 2);
  unsigned short* wbhh1 = (unsigned short*)take((size_t)N3 * H * 2);
  float*          gx    = (float*)take(MT * (size_t)N3 * 4);
  float*          hf    = (float*)take(2 * (size_t)B * H * 4);
  unsigned short* hb    = (unsigned short*)take(2 * (size_t)B * H * 2);

  const size_t BH = (size_t)B * H;

  // --- precision conversion (fp32 -> bf16) ---
  cvt_f32_bf16_kernel<<<2048, 256, 0, stream>>>(x, xb, MT * D);
  cvt_f32_bf16_kernel<<<1024, 256, 0, stream>>>(wih0, wbih0, (size_t)N3 * D);
  cvt_f32_bf16_kernel<<<1024, 256, 0, stream>>>(whh0, wbhh0, (size_t)N3 * H);
  cvt_f32_bf16_kernel<<<1024, 256, 0, stream>>>(wih1, wbih1, (size_t)N3 * H);
  cvt_f32_bf16_kernel<<<1024, 256, 0, stream>>>(whh1, wbhh1, (size_t)N3 * H);

  const dim3 gemmGrid((unsigned)(MT / BM), (unsigned)(N3 / BN));

  // ====================== Layer 0 ======================
  gemm_bias_kernel<<<gemmGrid, 256, 0, stream>>>(xb, wbih0, bih0, gx, (int)MT, N3, D);
  zero_h_kernel<<<(unsigned)(BH / 256), 256, 0, stream>>>(hf, hb, (int)BH);
  for (int t = 0; t < T; ++t) {
    const int pi = t & 1, po = (t + 1) & 1;
    gru_step_kernel<<<dim3(H / 16), 128, 0, stream>>>(
        hb + pi * BH, wbhh0, gx, bhh0,
        hf + pi * BH, hf + po * BH, hb + po * BH,
        y0b, (float*)nullptr, t, T, H);
  }
  // T even -> final h in parity 0
  copy_f32_kernel<<<(unsigned)(BH / 256), 256, 0, stream>>>(hf, out + MT * H, (int)BH);

  // ====================== Layer 1 ======================
  gemm_bias_kernel<<<gemmGrid, 256, 0, stream>>>(y0b, wbih1, bih1, gx, (int)MT, N3, H);
  zero_h_kernel<<<(unsigned)(BH / 256), 256, 0, stream>>>(hf, hb, (int)BH);
  for (int t = 0; t < T; ++t) {
    const int pi = t & 1, po = (t + 1) & 1;
    gru_step_kernel<<<dim3(H / 16), 128, 0, stream>>>(
        hb + pi * BH, wbhh1, gx, bhh1,
        hf + pi * BH, hf + po * BH, hb + po * BH,
        (unsigned short*)nullptr, out, t, T, H);
  }
  copy_f32_kernel<<<(unsigned)(BH / 256), 256, 0, stream>>>(hf, out + MT * H + BH, (int)BH);
}
